// Pro2Mamba_75170517615294
// MI455X (gfx1250) — matrix operand order repaired and checked by reading, not compiled, not run on hardware
//
#include <hip/hip_runtime.h>
#include <hip/hip_bf16.h>

#define BB 4
#define TTT 256
#define NNN 49
#define DDD 768
#define BT (BB*TTT)          // 1024
#define BTN (BB*TTT*NNN)     // 50176
#define DIN 1536
#define DSTATE 16
#define DTR 48
#define DCONV 4
#define CCLS 22

typedef __bf16 bf16_t;
typedef bf16_t v16bf __attribute__((ext_vector_type(16)));
typedef bf16_t v4bf  __attribute__((ext_vector_type(4)));
typedef float  v8f   __attribute__((ext_vector_type(8)));

#define ASTR 40   // LDS row stride (bf16): 32 data + 8 skew -> 20-bank stagger, conflict-free
#define NCOL 128  // block column tile

__device__ __forceinline__ float sigf(float x){ return 1.0f/(1.0f+__expf(-x)); }

// ---------------- small utility kernels ----------------

// transpose-convert: src fp32 (K,N) row-major -> dst bf16 (N,K) row-major
__global__ void k_cvtT(const float* __restrict__ s, bf16_t* __restrict__ d, int K, int N){
  int i = blockIdx.x*256 + threadIdx.x;
  if(i < N*K){
    int n = i / K, k = i - n*K;
    d[i] = (bf16_t)s[(size_t)k*N + n];
  }
}

// l2-normalize proto rows -> bf16 (22 x 768) row-major  (== B^T for the sim GEMMs)
__global__ void k_proto(const float* __restrict__ proto, bf16_t* __restrict__ pn){
  __shared__ float red[256];
  int c = blockIdx.x; int tid = threadIdx.x;
  float s=0.f;
  for(int k=tid;k<DDD;k+=256){ float v=proto[c*DDD+k]; s+=v*v; }
  red[tid]=s; __syncthreads();
  for(int off=128;off>0;off>>=1){ if(tid<off) red[tid]+=red[tid+off]; __syncthreads(); }
  float inv = 1.0f/fmaxf(sqrtf(red[0]), 1e-12f);
  for(int k=tid;k<DDD;k+=256) pn[c*DDD + k] = (bf16_t)(proto[c*DDD+k]*inv);
}

__global__ void k_token_mean(const float* __restrict__ tok, float* __restrict__ xm){
  int d = blockIdx.y*256 + threadIdx.x;
  int row = blockIdx.x;
  const float* p = tok + (size_t)row*NNN*DDD + d;
  float s=0.f;
  for(int n=0;n<NNN;++n) s += p[(size_t)n*DDD];
  xm[(size_t)row*DDD + d] = s * (1.0f/NNN);
}

// layernorm over D=768 (one block per row); optional per-row gate multiplier
__global__ void k_ln(const float* __restrict__ src, const float* __restrict__ gate,
                     const float* __restrict__ g, const float* __restrict__ b,
                     float* __restrict__ dst){
  __shared__ float rs[256], rq[256];
  int row = blockIdx.x, tid = threadIdx.x;
  float gv = gate ? gate[row] : 1.0f;
  float v0 = src[(size_t)row*DDD + tid      ]*gv;
  float v1 = src[(size_t)row*DDD + tid + 256]*gv;
  float v2 = src[(size_t)row*DDD + tid + 512]*gv;
  rs[tid]=v0+v1+v2; rq[tid]=v0*v0+v1*v1+v2*v2; __syncthreads();
  for(int off=128;off>0;off>>=1){ if(tid<off){rs[tid]+=rs[tid+off]; rq[tid]+=rq[tid+off];} __syncthreads(); }
  float m = rs[0]*(1.0f/DDD);
  float var = rq[0]*(1.0f/DDD) - m*m;
  float inv = rsqrtf(var + 1e-5f);
  dst[(size_t)row*DDD+tid      ] = (v0-m)*inv*g[tid      ] + b[tid      ];
  dst[(size_t)row*DDD+tid+256  ] = (v1-m)*inv*g[tid+256  ] + b[tid+256  ];
  dst[(size_t)row*DDD+tid+512  ] = (v2-m)*inv*g[tid+512  ] + b[tid+512  ];
}

// temporal-evidence gate: per (b,t) diff magnitudes -> stats -> tiny MLP -> sigmoid
__global__ void k_gate(const float* __restrict__ tok,
                       const float* __restrict__ w1, const float* __restrict__ b1,
                       const float* __restrict__ w2, const float* __restrict__ b2,
                       float* __restrict__ gate){
  __shared__ float red[3][256];
  __shared__ float mag[NNN];
  int row = blockIdx.x; int tid = threadIdx.x;
  int b = row / TTT, t = row % TTT;
  const int dil[3] = {1,6,12};
  for(int n=0;n<NNN;++n){
    float acc0=0.f, acc1=0.f, acc2=0.f;
    for(int d=tid; d<DDD; d+=256){
      float xa = tok[((size_t)((b*TTT+t)*NNN + n))*DDD + d];
      int s0 = t-dil[0]; if(s0<0) s0=0;
      int s1 = t-dil[1]; if(s1<0) s1=0;
      int s2 = t-dil[2]; if(s2<0) s2=0;
      float d0 = xa - tok[((size_t)((b*TTT+s0)*NNN + n))*DDD + d];
      float d1 = xa - tok[((size_t)((b*TTT+s1)*NNN + n))*DDD + d];
      float d2 = xa - tok[((size_t)((b*TTT+s2)*NNN + n))*DDD + d];
      acc0+=d0*d0; acc1+=d1*d1; acc2+=d2*d2;
    }
    red[0][tid]=acc0; red[1][tid]=acc1; red[2][tid]=acc2; __syncthreads();
    for(int off=128;off>0;off>>=1){
      if(tid<off){ red[0][tid]+=red[0][tid+off]; red[1][tid]+=red[1][tid+off]; red[2][tid]+=red[2][tid+off]; }
      __syncthreads();
    }
    if(tid==0) mag[n]=(sqrtf(red[0][0])+sqrtf(red[1][0])+sqrtf(red[2][0]))*(1.0f/3.0f);
    __syncthreads();
  }
  if(tid==0){
    float m=0.f; for(int n=0;n<NNN;++n) m+=mag[n]; m*=(1.0f/NNN);
    float v=0.f; for(int n=0;n<NNN;++n){float dd=mag[n]-m; v+=dd*dd;} v*=(1.0f/(NNN-1));
    float gg=0.f;
    for(int j=0;j<16;++j){
      float h = m*w1[j] + v*w1[16+j] + b1[j];
      h = fmaxf(h,0.f);
      gg += h*w2[j];
    }
    gg += b2[0];
    gate[row] = sigf(gg);
  }
}

// ---------------- shared WMMA tile helpers ----------------
// A frag: lane m = lane&15 holds row (wave*16+m); k chunks {0..7}+khalf, {16..23}+khalf
__device__ __forceinline__ v16bf load_a_frag(const bf16_t* As, int wave, int lane){
  v16bf af;
  int m = lane&15, khalf=(lane>>4)<<3;
  const bf16_t* ap = &As[(wave*16+m)*ASTR];
  #pragma unroll
  for(int e=0;e<16;++e){
    int k=(e&7)+((e>>3)<<4)+khalf;
    af[e]=ap[k];
  }
  return af;
}
// B frag (B^T staged N-major): lane n = lane&15; k chunks {0..7}+kb, {16..23}+kb, kb=(lane>>4)*8
__device__ __forceinline__ v16bf load_b_frag(const bf16_t* Bs, int cg, int lane){
  v16bf bfr;
  int n_=lane&15, kb=(lane>>4)<<3;
  const bf16_t* bp = &Bs[(cg*16+n_)*ASTR + kb];
  #pragma unroll
  for(int e=0;e<16;++e) bfr[e]=bp[(e&7)+((e>>3)<<4)];
  return bfr;
}

// ---------------- GEMM-1: relu(diffcat @ W1 + b1), mean over N via atomics ----------------
// grid (BTN/64, 768/128), 128 threads (4 waves, each 16 rows x 128 cols)
__global__ void k_fuse_gemm1(const float* __restrict__ tok, const bf16_t* __restrict__ W1T,
                             const float* __restrict__ fbias, float* __restrict__ hmean){
  __shared__ bf16_t As[64*ASTR];
  __shared__ bf16_t Bs[NCOL*ASTR];
  const int dil[3]={1,6,12};
  int tid=threadIdx.x, lane=tid&31, wave=tid>>5;
  int rowBase = blockIdx.x*64, colBase = blockIdx.y*NCOL;
  v8f acc[8];
  #pragma unroll
  for(int i=0;i<8;++i){ acc[i][0]=0.f;acc[i][1]=0.f;acc[i][2]=0.f;acc[i][3]=0.f;
                        acc[i][4]=0.f;acc[i][5]=0.f;acc[i][6]=0.f;acc[i][7]=0.f; }

  for(int k0=0;k0<3*DDD;k0+=32){
    int seg = k0/DDD;
    int cbase = k0 - seg*DDD;
    int dl = dil[seg];
    // stage A (64x32): diffs computed on the fly, fp32 -> bf16
    for(int q=tid;q<512;q+=128){
      int r = q>>3, cl=(q&7)*4;
      int grow = rowBase + r;
      int b = grow/(TTT*NNN); int rem = grow - b*(TTT*NNN);
      int t = rem/NNN; int n = rem - t*NNN;
      int st = t - dl; if(st<0) st=0;
      float4 xa = *(const float4*)(tok + ((size_t)((b*TTT+t )*NNN+n))*DDD + cbase + cl);
      float4 xs = *(const float4*)(tok + ((size_t)((b*TTT+st)*NNN+n))*DDD + cbase + cl);
      v4bf w; w[0]=(bf16_t)(xa.x-xs.x); w[1]=(bf16_t)(xa.y-xs.y);
              w[2]=(bf16_t)(xa.z-xs.z); w[3]=(bf16_t)(xa.w-xs.w);
      *(v4bf*)(&As[r*ASTR+cl]) = w;
    }
    // stage B (128 cols x 32 k) from pre-transposed bf16 W1^T (768 x 2304)
    for(int q=tid;q<1024;q+=128){
      int nl=q>>3, cl=(q&7)*4;
      v4bf w = *(const v4bf*)(W1T + (size_t)(colBase+nl)*(3*DDD) + k0 + cl);
      *(v4bf*)(&Bs[nl*ASTR+cl]) = w;
    }
    __syncthreads();
    v16bf af = load_a_frag(As, wave, lane);
    #pragma unroll
    for(int cg=0;cg<8;++cg){
      v16bf bfr = load_b_frag(Bs, cg, lane);
      acc[cg]=__builtin_amdgcn_wmma_f32_16x16x32_bf16(false,af,false,bfr,(short)0,acc[cg],false,false);
    }
    __syncthreads();
  }
  // epilogue: bias + relu, accumulate mean over N into hmean (per (b,t))
  int m16base = (lane>>4)<<3;
  int col0 = lane&15;
  for(int cg=0;cg<8;++cg){
    #pragma unroll
    for(int j=0;j<8;++j){
      int grow = rowBase + wave*16 + j + m16base;
      int gcol = colBase + cg*16 + col0;
      int bt = grow / NNN;
      float v = acc[cg][j] + fbias[gcol];
      v = fmaxf(v,0.f)*(1.0f/NNN);
      atomicAdd(&hmean[(size_t)bt*DDD + gcol], v);
    }
  }
}

// ---------------- generic WMMA GEMM: C = epilogue(A@B [+bias]) [+addsrc] ----------------
// A fp32 (M,K) stride lda; BT bf16 = B^T (N,K) stride ldb; C fp32 stride ldc.
// act: 0=none 1=relu 2=softplus
__global__ void k_gemm(const float* __restrict__ A, int lda,
                       const bf16_t* __restrict__ BwT, int ldb,
                       const float* __restrict__ bias,
                       const float* __restrict__ addsrc,
                       float* __restrict__ C, int ldc,
                       int M, int N, int K, int act){
  __shared__ bf16_t As[64*ASTR];
  __shared__ bf16_t Bs[NCOL*ASTR];
  int tid=threadIdx.x, lane=tid&31, wave=tid>>5;
  int rowBase=blockIdx.x*64, colBase=blockIdx.y*NCOL;
  v8f acc[8];
  #pragma unroll
  for(int i=0;i<8;++i){ acc[i][0]=0.f;acc[i][1]=0.f;acc[i][2]=0.f;acc[i][3]=0.f;
                        acc[i][4]=0.f;acc[i][5]=0.f;acc[i][6]=0.f;acc[i][7]=0.f; }

  for(int k0=0;k0<K;k0+=32){
    // stage A (64 x 32)
    for(int q=tid;q<512;q+=128){
      int r=q>>3, cl=(q&7)*4;
      int grow=rowBase+r;
      if(grow<M && (k0+cl+3)<K){
        float4 xv = *(const float4*)(A + (size_t)grow*lda + k0 + cl);
        v4bf w; w[0]=(bf16_t)xv.x; w[1]=(bf16_t)xv.y; w[2]=(bf16_t)xv.z; w[3]=(bf16_t)xv.w;
        *(v4bf*)(&As[r*ASTR+cl]) = w;
      } else {
        #pragma unroll
        for(int e=0;e<4;++e){
          int k=k0+cl+e;
          float v = (grow<M && k<K) ? A[(size_t)grow*lda + k] : 0.f;
          As[r*ASTR+cl+e]=(bf16_t)v;
        }
      }
    }
    // stage B^T (128 cols x 32 k)
    for(int q=tid;q<1024;q+=128){
      int nl=q>>3, cl=(q&7)*4;
      int n=colBase+nl;
      if(n<N && (k0+cl+3)<K){
        v4bf w = *(const v4bf*)(BwT + (size_t)n*ldb + k0 + cl);
        *(v4bf*)(&Bs[nl*ASTR+cl]) = w;
      } else {
        #pragma unroll
        for(int e=0;e<4;++e){
          int k=k0+cl+e;
          Bs[nl*ASTR+cl+e] = (n<N && k<K) ? BwT[(size_t)n*ldb + k] : (bf16_t)0.0f;
        }
      }
    }
    __syncthreads();
    v16bf af = load_a_frag(As, wave, lane);
    #pragma unroll
    for(int cg=0;cg<8;++cg){
      v16bf bfr = load_b_frag(Bs, cg, lane);
      acc[cg]=__builtin_amdgcn_wmma_f32_16x16x32_bf16(false,af,false,bfr,(short)0,acc[cg],false,false);
    }
    __syncthreads();
  }
  int m16base=(lane>>4)<<3, col0=lane&15;
  for(int cg=0;cg<8;++cg){
    #pragma unroll
    for(int j=0;j<8;++j){
      int grow=rowBase+wave*16+j+m16base;
      int gcol=colBase+cg*16+col0;
      if(grow<M && gcol<N){
        float v=acc[cg][j];
        if(bias) v+=bias[gcol];
        if(act==1) v=fmaxf(v,0.f);
        else if(act==2) v=(v>20.f)? v : log1pf(__expf(v));
        if(addsrc) v+=addsrc[(size_t)grow*ldc+gcol];
        C[(size_t)grow*ldc+gcol]=v;
      }
    }
  }
}

// ---------------- mamba depthwise conv + silu ----------------
__global__ void k_conv(const float* __restrict__ xz, const float* __restrict__ cw,
                       const float* __restrict__ cb, float* __restrict__ xa){
  int idx = blockIdx.x*256+threadIdx.x;
  if(idx >= BT*DIN) return;
  int row = idx/DIN, d = idx - row*DIN;
  int b = row/TTT, t = row - b*TTT;
  float acc = cb[d];
  #pragma unroll
  for(int k=0;k<DCONV;++k){
    int tt = t + k - (DCONV-1);
    if(tt>=0) acc += xz[(size_t)(b*TTT+tt)*(2*DIN) + d] * cw[d*DCONV+k];
  }
  xa[(size_t)idx] = acc * sigf(acc);
}

// ---------------- mamba selective scan (sequential over T) ----------------
__global__ void k_scan(const float* __restrict__ dtv, const float* __restrict__ xa,
                       const float* __restrict__ dbl, const float* __restrict__ xz,
                       const float* __restrict__ A_log, const float* __restrict__ Dp,
                       float* __restrict__ ya){
  int d = blockIdx.y*256 + threadIdx.x;
  int b = blockIdx.x;
  float Ae[DSTATE];
  #pragma unroll
  for(int s=0;s<DSTATE;++s) Ae[s] = -__expf(A_log[d*DSTATE+s]);
  float Dd = Dp[d];
  float h[DSTATE];
  #pragma unroll
  for(int s=0;s<DSTATE;++s) h[s]=0.f;
  for(int t=0;t<TTT;++t){
    int row = b*TTT+t;
    float dt = dtv[(size_t)row*DIN + d];
    float xv = xa [(size_t)row*DIN + d];
    float zv = xz [(size_t)row*(2*DIN) + DIN + d];
    float y = 0.f;
    #pragma unroll
    for(int s=0;s<DSTATE;++s){
      float Bv = dbl[row*80 + DTR + s];
      float Cv = dbl[row*80 + DTR + DSTATE + s];
      float dA = __expf(dt*Ae[s]);
      h[s] = dA*h[s] + dt*Bv*xv;
      y += h[s]*Cv;
    }
    y += xv*Dd;
    y *= zv*sigf(zv);
    ya[(size_t)row*DIN + d] = y;
  }
}

// ---------------- final head: cosine sims combine ----------------
__global__ void k_final(const float* __restrict__ z, const float* __restrict__ zd,
                        const float* __restrict__ s1, const float* __restrict__ s2,
                        const float* __restrict__ scale, float* __restrict__ out){
  __shared__ float r1[256], r2[256];
  int row=blockIdx.x, tid=threadIdx.x;
  float a0=z [(size_t)row*DDD+tid], a1=z [(size_t)row*DDD+tid+256], a2=z [(size_t)row*DDD+tid+512];
  float b0=zd[(size_t)row*DDD+tid], b1=zd[(size_t)row*DDD+tid+256], b2=zd[(size_t)row*DDD+tid+512];
  r1[tid]=a0*a0+a1*a1+a2*a2; r2[tid]=b0*b0+b1*b1+b2*b2; __syncthreads();
  for(int off=128;off>0;off>>=1){ if(tid<off){r1[tid]+=r1[tid+off]; r2[tid]+=r2[tid+off];} __syncthreads(); }
  float inz  = 1.0f/fmaxf(sqrtf(r1[0]),1e-12f);
  float inzd = 1.0f/fmaxf(sqrtf(r2[0]),1e-12f);
  float sc = scale[0];
  if(tid<CCLS)
    out[row*CCLS+tid] = (s1[row*CCLS+tid]*inz + 0.5f*s2[row*CCLS+tid]*inzd)*sc;
}

// ---------------- host orchestration ----------------
extern "C" void kernel_launch(void* const* d_in, const int* in_sizes, int n_in,
                              void* d_out, int out_size, void* d_ws, size_t ws_size,
                              hipStream_t stream){
  (void)in_sizes; (void)n_in; (void)out_size; (void)ws_size;
  const float* tokens = (const float*)d_in[0];
  const float* ln_g = (const float*)d_in[1];
  const float* ln_b = (const float*)d_in[2];
  const float* fw1  = (const float*)d_in[3];
  const float* fb1  = (const float*)d_in[4];
  const float* fw2  = (const float*)d_in[5];
  const float* fb2  = (const float*)d_in[6];
  const float* gw1  = (const float*)d_in[7];
  const float* gb1  = (const float*)d_in[8];
  const float* gw2  = (const float*)d_in[9];
  const float* gb2  = (const float*)d_in[10];
  const float* m_in_w[2] = {(const float*)d_in[11], (const float*)d_in[20]};
  const float* m_cw[2]   = {(const float*)d_in[12], (const float*)d_in[21]};
  const float* m_cb[2]   = {(const float*)d_in[13], (const float*)d_in[22]};
  const float* m_xp[2]   = {(const float*)d_in[14], (const float*)d_in[23]};
  const float* m_dtw[2]  = {(const float*)d_in[15], (const float*)d_in[24]};
  const float* m_dtb[2]  = {(const float*)d_in[16], (const float*)d_in[25]};
  const float* m_alog[2] = {(const float*)d_in[17], (const float*)d_in[26]};
  const float* m_D[2]    = {(const float*)d_in[18], (const float*)d_in[27]};
  const float* m_ow[2]   = {(const float*)d_in[19], (const float*)d_in[28]};
  const float* fus_w = (const float*)d_in[29];
  const float* fus_b = (const float*)d_in[30];
  const float* proto = (const float*)d_in[31];
  const float* ev_w1 = (const float*)d_in[32];
  const float* ev_b1 = (const float*)d_in[33];
  const float* ev_w2 = (const float*)d_in[34];
  const float* ev_b2 = (const float*)d_in[35];
  const float* scale = (const float*)d_in[36];

  char* wsp = (char*)d_ws; size_t off=0;
  auto carve = [&](size_t bytes)->char*{
    char* p = wsp+off; off = (off + bytes + 255) & ~(size_t)255; return p;
  };
  // bf16 transposed weight copies (B^T = (N,K) row-major)
  bf16_t* W1T   = (bf16_t*)carve((size_t)768*2304*2);
  bf16_t* W2T   = (bf16_t*)carve((size_t)768*768*2);
  bf16_t *inWT[2], *xpT[2], *dtwT[2], *owT[2];
  for(int i=0;i<2;++i){
    inWT[i] = (bf16_t*)carve((size_t)3072*768*2);
    xpT[i]  = (bf16_t*)carve((size_t)80*1536*2);
    dtwT[i] = (bf16_t*)carve((size_t)1536*48*2);
    owT[i]  = (bf16_t*)carve((size_t)768*1536*2);
  }
  bf16_t* fusT = (bf16_t*)carve((size_t)768*1536*2);
  bf16_t* pnb  = (bf16_t*)carve((size_t)CCLS*768*2);
  bf16_t* ev1T = (bf16_t*)carve((size_t)768*768*2);
  bf16_t* ev2T = (bf16_t*)carve((size_t)768*768*2);
  // fp32 activations
  float* xmean  = (float*)carve((size_t)BT*DDD*4);   // reused as fused-mean
  float* xlong  = (float*)carve((size_t)BT*DDD*4);
  float* gateb  = (float*)carve((size_t)BT*4);
  float* hmean  = (float*)carve((size_t)BT*DDD*4);
  float* xtrans = (float*)carve((size_t)BT*DDD*4);
  float* xzb    = (float*)carve((size_t)BT*2*DIN*4);
  float* xab    = (float*)carve((size_t)BT*DIN*4);
  float* dblb   = (float*)carve((size_t)BT*80*4);
  float* dtvb   = (float*)carve((size_t)BT*DIN*4);
  float* yab    = (float*)carve((size_t)BT*DIN*4);
  float* zcat   = (float*)carve((size_t)BT*DIN*4);
  float* zbuf   = (float*)carve((size_t)BT*DDD*4);
  float* ebuf   = (float*)carve((size_t)BT*DDD*4);
  float* zdbuf  = (float*)carve((size_t)BT*DDD*4);
  float* s1b    = (float*)carve((size_t)BT*CCLS*4);
  float* s2b    = (float*)carve((size_t)BT*CCLS*4);

  auto cvtT = [&](const float* s, bf16_t* d, int K, int N){
    int cnt = K*N;
    k_cvtT<<<(cnt+255)/256, 256, 0, stream>>>(s, d, K, N);
  };
  auto gemm = [&](const float* A,int lda,const bf16_t* BwT,int ldb,const float* bias,
                  const float* add,float* C,int ldc,int M,int N,int K,int act){
    dim3 g((M+63)/64, (N+NCOL-1)/NCOL);
    k_gemm<<<g, 128, 0, stream>>>(A,lda,BwT,ldb,bias,add,C,ldc,M,N,K,act);
  };

  // 1. weight conversions (transposed)
  cvtT(fw1, W1T, 2304, 768);
  cvtT(fw2, W2T, 768, 768);
  for(int i=0;i<2;++i){
    cvtT(m_in_w[i], inWT[i], 768, 3072);
    cvtT(m_xp[i],   xpT[i],  1536, 80);
    cvtT(m_dtw[i],  dtwT[i], 48, 1536);
    cvtT(m_ow[i],   owT[i],  1536, 768);
  }
  cvtT(fus_w, fusT, 1536, 768);
  cvtT(ev_w1, ev1T, 768, 768);   // ctx rows (768..789) multiply zeros -> dropped
  cvtT(ev_w2, ev2T, 768, 768);
  k_proto<<<CCLS, 256, 0, stream>>>(proto, pnb);

  // 2. x_long = LN(tokens.mean(n))
  k_token_mean<<<dim3(BT,3), 256, 0, stream>>>(tokens, xmean);
  k_ln<<<BT, 256, 0, stream>>>(xmean, nullptr, ln_g, ln_b, xlong);

  // 3. temporal gate
  k_gate<<<BT, 256, 0, stream>>>(tokens, gw1, gb1, gw2, gb2, gateb);

  // 4. GEMM-1 with fused relu + mean-over-N accumulation
  hipMemsetAsync(hmean, 0, (size_t)BT*DDD*4, stream);
  k_fuse_gemm1<<<dim3(BTN/64, DDD/NCOL), 128, 0, stream>>>(tokens, W1T, fb1, hmean);

  // 5. fused.mean = hmean @ W2 + b2 ; x_trans = LN(fused.mean * gate)
  gemm(hmean,DDD, W2T,DDD, fb2, nullptr, xmean,DDD, BT,DDD,DDD, 0);
  k_ln<<<BT, 256, 0, stream>>>(xmean, gateb, ln_g, ln_b, xtrans);

  // 6. two mamba blocks
  for(int i=0;i<2;++i){
    const float* xin = (i==0) ? xlong : xtrans;
    gemm(xin,DDD, inWT[i],DDD, nullptr, nullptr, xzb,2*DIN, BT,2*DIN,DDD, 0);
    k_conv<<<(BT*DIN)/256, 256, 0, stream>>>(xzb, m_cw[i], m_cb[i], xab);
    gemm(xab,DIN, xpT[i],DIN, nullptr, nullptr, dblb,80, BT,80,DIN, 0);
    gemm(dblb,80, dtwT[i],DTR, m_dtb[i], nullptr, dtvb,DIN, BT,DIN,DTR, 2); // softplus
    k_scan<<<dim3(BB, DIN/256), 256, 0, stream>>>(dtvb, xab, dblb, xzb,
                                                  m_alog[i], m_D[i], yab);
    gemm(yab,DIN, owT[i],DIN, nullptr, nullptr, zcat + i*DDD, DIN, BT,DDD,DIN, 0);
  }

  // 7. fusion + head
  gemm(zcat,DIN, fusT,DIN, fus_b, nullptr, zbuf,DDD, BT,DDD,DIN, 0);
  gemm(zbuf,DDD, pnb,DDD, nullptr, nullptr, s1b,CCLS, BT,CCLS,DDD, 0);
  gemm(zbuf,DDD, ev1T,DDD, ev_b1, nullptr, ebuf,DDD, BT,DDD,DDD, 1);          // relu
  gemm(ebuf,DDD, ev2T,DDD, ev_b2, zbuf, zdbuf,DDD, BT,DDD,DDD, 0);            // z + delta
  gemm(zdbuf,DDD, pnb,DDD, nullptr, nullptr, s2b,CCLS, BT,CCLS,DDD, 0);
  k_final<<<BT, 256, 0, stream>>>(zbuf, zdbuf, s1b, s2b, scale, (float*)d_out);
}
